// TaskAssignmentNet_36850819400097
// MI455X (gfx1250) — compile-verified, matrix operand order checked by reading, not run on hardware
//
#include <hip/hip_runtime.h>
#include <hip/hip_bf16.h>
#include <math.h>

// ---------------- model constants ----------------
static const int kNT  = 100000;      // tasks
static const int kND  = 50000;       // data nodes
static const int kNV  = 8;           // devices
static const int kEDT = 300000, kEVT = 300000, kETT = 200000;

typedef __attribute__((ext_vector_type(16))) __bf16 v16bf;
typedef __attribute__((ext_vector_type(8)))  __bf16 v8bf;
typedef __attribute__((ext_vector_type(8)))  float  v8f;

__device__ __forceinline__ unsigned short f2bf(float f) {
  unsigned u = __float_as_uint(f);
  u += 0x7FFFu + ((u >> 16) & 1u);            // round-to-nearest-even
  return (unsigned short)(u >> 16);
}
// monotonic float<->uint mapping for atomicMax-based segment_max
__device__ __forceinline__ unsigned f2ord(float f) {
  unsigned u = __float_as_uint(f);
  return (u & 0x80000000u) ? ~u : (u | 0x80000000u);
}
__device__ __forceinline__ float ord2f(unsigned k) {
  unsigned u = (k & 0x80000000u) ? (k ^ 0x80000000u) : ~k;
  return __uint_as_float(u);
}

// ---------------- utility kernels ----------------
__global__ void k_fill_f32(float* p, float v, long long n) {
  long long i = (long long)blockIdx.x * blockDim.x + threadIdx.x;
  if (i < n) p[i] = v;
}
__global__ void k_fill_u32(unsigned* p, unsigned v, long long n) {
  long long i = (long long)blockIdx.x * blockDim.x + threadIdx.x;
  if (i < n) p[i] = v;
}
// fp32 [rs,cs] -> bf16 [rd,cd], zero padded
__global__ void k_cvt_bf16_pad(const float* __restrict__ src, unsigned short* __restrict__ dst,
                               int rs, int cs, int rd, int cd) {
  long long i = (long long)blockIdx.x * blockDim.x + threadIdx.x;
  long long n = (long long)rd * cd;
  if (i >= n) return;
  int r = (int)(i / cd), c = (int)(i % cd);
  float f = (r < rs && c < cs) ? src[(long long)r * cs + c] : 0.f;
  dst[i] = f2bf(f);
}
// W[K,N] f32 row-major -> WT[N,Kpad] bf16 (transposed, K zero-padded)
__global__ void k_wt_bf16(const float* __restrict__ W, unsigned short* __restrict__ WT,
                          int K, int N, int Kpad) {
  long long i = (long long)blockIdx.x * blockDim.x + threadIdx.x;
  long long n = (long long)N * Kpad;
  if (i >= n) return;
  int nn = (int)(i / Kpad), k = (int)(i % Kpad);
  WT[i] = f2bf(k < K ? W[(long long)k * N + nn] : 0.f);
}
// fold[f,2] = sum_c W[f, h*64+c] * att[h*64+c]
__global__ void k_fold_att(const float* __restrict__ W, const float* __restrict__ att,
                           float* __restrict__ fold, int f) {
  int i = blockIdx.x * blockDim.x + threadIdx.x;
  if (i >= f * 2) return;
  int r = i >> 1, h = i & 1;
  float s = 0.f;
  #pragma unroll
  for (int c = 0; c < 64; c++) s += W[r * 128 + h * 64 + c] * att[h * 64 + c];
  fold[i] = s;
}
// out[N,2] = X[N,f] @ fold[f,2]   (a_src / a_dst without full xd GEMM)
__global__ void k_att_nodes(const float* __restrict__ X, const float* __restrict__ fold,
                            float* __restrict__ out, int N, int f) {
  int i = blockIdx.x * blockDim.x + threadIdx.x;
  if (i >= N) return;
  float s0 = 0.f, s1 = 0.f;
  for (int k = 0; k < f; k++) {
    float x = X[(long long)i * f + k];
    s0 += x * fold[2 * k]; s1 += x * fold[2 * k + 1];
  }
  out[2 * i] = s0; out[2 * i + 1] = s1;
}
// a_e[E,2] = ea[E,8] @ wef[8,2]
__global__ void k_edge_ae(const float* __restrict__ ea, const float* __restrict__ wef,
                          float* __restrict__ ae, int E) {
  int e = blockIdx.x * blockDim.x + threadIdx.x;
  if (e >= E) return;
  float s0 = 0.f, s1 = 0.f;
  #pragma unroll
  for (int k = 0; k < 8; k++) {
    float x = ea[(long long)e * 8 + k];
    s0 += x * wef[2 * k]; s1 += x * wef[2 * k + 1];
  }
  ae[2 * e] = s0; ae[2 * e + 1] = s1;
}
// column sums of ea_tt (for self-loop attrs): grid-stride + LDS partials
__global__ void k_colsum8(const float* __restrict__ ea, float* __restrict__ sum8, int E) {
  __shared__ float ls[8];
  if (threadIdx.x < 8) ls[threadIdx.x] = 0.f;
  __syncthreads();
  float acc[8] = {0, 0, 0, 0, 0, 0, 0, 0};
  for (long long e = (long long)blockIdx.x * blockDim.x + threadIdx.x; e < E;
       e += (long long)gridDim.x * blockDim.x) {
    #pragma unroll
    for (int k = 0; k < 8; k++) acc[k] += ea[e * 8 + k];
  }
  #pragma unroll
  for (int k = 0; k < 8; k++) atomicAdd(&ls[k], acc[k]);
  __syncthreads();
  if (threadIdx.x < 8) atomicAdd(&sum8[threadIdx.x], ls[threadIdx.x]);
}
__global__ void k_loop_ae(const float* sum8, const float* wef, float* aloop, float invE) {
  if (blockIdx.x == 0 && threadIdx.x == 0) {
    float s0 = 0.f, s1 = 0.f;
    for (int k = 0; k < 8; k++) {
      float m = sum8[k] * invE;
      s0 += m * wef[2 * k]; s1 += m * wef[2 * k + 1];
    }
    aloop[0] = s0; aloop[1] = s1;
  }
}
// edge pass 1: alpha = lrelu0.2(a_src+a_dst+a_e); segment max via atomicMax
__global__ void k_edge_alpha(const int* __restrict__ src, const int* __restrict__ dst,
                             const float* __restrict__ asrc, const float* __restrict__ adst,
                             const float* __restrict__ ae, const float* __restrict__ aloop,
                             float* __restrict__ alpha, unsigned* __restrict__ amax,
                             int E_idx, int E_tot) {
  int t = blockIdx.x * blockDim.x + threadIdx.x;
  if (t >= 2 * E_tot) return;
  int e = t >> 1, h = t & 1;
  int s, d; float av;
  if (e < E_idx) { s = src[e]; d = dst[e]; av = ae[t]; }
  else           { s = e - E_idx; d = s;   av = aloop[h]; }   // self loop
  float a = asrc[2 * s + h] + adst[2 * d + h] + av;
  a = a > 0.f ? a : 0.2f * a;
  alpha[t] = a;
  atomicMax(&amax[2 * d + h], f2ord(a));
}
// edge pass 2: ex = exp(alpha - amax[dst]); den += ex
__global__ void k_edge_exp(const int* __restrict__ dst, float* __restrict__ alpha,
                           const unsigned* __restrict__ amax, float* __restrict__ den,
                           int E_idx, int E_tot) {
  int t = blockIdx.x * blockDim.x + threadIdx.x;
  if (t >= 2 * E_tot) return;
  int e = t >> 1, h = t & 1;
  int d = (e < E_idx) ? dst[e] : (e - E_idx);
  float ex = expf(alpha[t] - ord2f(amax[2 * d + h]));
  alpha[t] = ex;
  atomicAdd(&den[2 * d + h], ex);
}
// edge pass 3: outacc[dst] += xs[src]*w  (32 lanes x float4 per edge)
__global__ void k_edge_scatter(const int* __restrict__ src, const int* __restrict__ dst,
                               const float* __restrict__ ex, const float* __restrict__ den,
                               const float* __restrict__ xs, float* __restrict__ outacc,
                               int E_idx, int E_tot) {
  long long t = (long long)blockIdx.x * blockDim.x + threadIdx.x;
  if (t >= (long long)E_tot * 32) return;
  int e = (int)(t >> 5), c4 = (int)(t & 31) * 4, h = c4 >> 6;
  int s, d;
  if (e < E_idx) { s = src[e]; d = dst[e]; }
  else           { s = e - E_idx; d = s; }
  float w = ex[2 * e + h] / (den[2 * d + h] + 1e-16f);
  const float4 x = *(const float4*)(xs + (long long)s * 128 + c4);
  float* o = outacc + (long long)d * 128 + c4;
  atomicAdd(o + 0, x.x * w); atomicAdd(o + 1, x.y * w);
  atomicAdd(o + 2, x.z * w); atomicAdd(o + 3, x.w * w);
}

// shared-memory tree reduction for LN (64 threads): returns mean & rstd via red[]
__device__ __forceinline__ void ln_stats(float v, int c, float* s1, float* s2, float* red) {
  s1[c] = v; s2[c] = v * v;
  __syncthreads();
  #pragma unroll
  for (int st = 32; st > 0; st >>= 1) {
    if (c < st) { s1[c] += s1[c + st]; s2[c] += s2[c + st]; }
    __syncthreads();
  }
  if (c == 0) {
    float m = s1[0] * (1.f / 64.f);
    float var = s2[0] * (1.f / 64.f) - m * m;
    red[0] = m; red[1] = 1.f / sqrtf(var + 1e-5f);
  }
  __syncthreads();
}

// per-row: head-mean + residual + bias -> LN -> lrelu0.01 -> xcat column block
__global__ void k_gat_epilogue(const float* __restrict__ outacc, const float* __restrict__ res,
                               const float* __restrict__ bias, const float* __restrict__ g,
                               const float* __restrict__ b, float* __restrict__ xcat, int col_off) {
  int row = blockIdx.x, c = threadIdx.x;
  __shared__ float s1[64], s2[64], red[2];
  float v = 0.5f * (outacc[(long long)row * 128 + c] + outacc[(long long)row * 128 + 64 + c])
            + res[(long long)row * 64 + c] + bias[c];
  ln_stats(v, c, s1, s2, red);
  float y = (v - red[0]) * red[1] * g[c] + b[c];
  y = y > 0.f ? y : 0.01f * y;
  xcat[(long long)row * 256 + col_off + c] = y;
}
__global__ void k_copy_tasks(const float* __restrict__ tx, float* __restrict__ xcat, long long n) {
  long long i = (long long)blockIdx.x * blockDim.x + threadIdx.x;
  if (i >= n) return;
  xcat[(long long)(i >> 6) * 256 + (i & 63)] = tx[i];
}

// ---------------- bf16 WMMA GEMM ----------------
// C[M,N](+col_off) = A[M,K]bf16 @ B (B given transposed: BT[N,K]bf16)
// One wave computes a 16x64 strip: 1 A fragment feeds 4 WMMAs per k-step (4x A reuse),
// fragment loads are software-pipelined one k-step ahead so s_wait overlaps the WMMAs.
__device__ __forceinline__ v16bf load_frag(const unsigned short* __restrict__ row, int k0, int half) {
  v8bf lo = *(const v8bf*)(row + k0 + half * 8);
  v8bf hi = *(const v8bf*)(row + k0 + 16 + half * 8);
  v16bf r;
  #pragma unroll
  for (int i = 0; i < 8; i++) { r[i] = lo[i]; r[i + 8] = hi[i]; }
  return r;
}
__global__ void k_gemm_bf16(const unsigned short* __restrict__ A, const unsigned short* __restrict__ BT,
                            float* __restrict__ Cm, int M, int K, int N, int ldc, int col_off) {
  int wave = (blockIdx.x * blockDim.x + threadIdx.x) >> 5;
  int lane = threadIdx.x & 31;
  int ngrp = N >> 6;                               // groups of 64 output columns
  int tiles = (M >> 4) * ngrp;
  if (wave >= tiles) return;                       // wave-uniform: EXEC all-ones inside
  int tm = wave / ngrp, tg = wave % ngrp;
  int mi = lane & 15, half = lane >> 4;
  const unsigned short* arow = A + (long long)(tm * 16 + mi) * K;
  const unsigned short* br0 = BT + (long long)((tg * 4 + 0) * 16 + mi) * K;
  const unsigned short* br1 = BT + (long long)((tg * 4 + 1) * 16 + mi) * K;
  const unsigned short* br2 = BT + (long long)((tg * 4 + 2) * 16 + mi) * K;
  const unsigned short* br3 = BT + (long long)((tg * 4 + 3) * 16 + mi) * K;
  v8f acc0 = {0.f,0.f,0.f,0.f,0.f,0.f,0.f,0.f}, acc1 = acc0, acc2 = acc0, acc3 = acc0;
  v16bf aC  = load_frag(arow, 0, half);
  v16bf bC0 = load_frag(br0, 0, half), bC1 = load_frag(br1, 0, half);
  v16bf bC2 = load_frag(br2, 0, half), bC3 = load_frag(br3, 0, half);
  for (int k0 = 0; k0 < K; k0 += 32) {
    int k1 = k0 + 32;
    v16bf aN = aC, bN0 = bC0, bN1 = bC1, bN2 = bC2, bN3 = bC3;
    if (k1 < K) {                                  // pipeline next fragments over the WMMAs
      __builtin_prefetch(arow + k1 + 32, 0, 3);    // keep streaming A in near cache
      aN  = load_frag(arow, k1, half);
      bN0 = load_frag(br0, k1, half); bN1 = load_frag(br1, k1, half);
      bN2 = load_frag(br2, k1, half); bN3 = load_frag(br3, k1, half);
    }
    acc0 = __builtin_amdgcn_wmma_f32_16x16x32_bf16(false, aC, false, bC0, (short)0, acc0, false, false);
    acc1 = __builtin_amdgcn_wmma_f32_16x16x32_bf16(false, aC, false, bC1, (short)0, acc1, false, false);
    acc2 = __builtin_amdgcn_wmma_f32_16x16x32_bf16(false, aC, false, bC2, (short)0, acc2, false, false);
    acc3 = __builtin_amdgcn_wmma_f32_16x16x32_bf16(false, aC, false, bC3, (short)0, acc3, false, false);
    aC = aN; bC0 = bN0; bC1 = bN1; bC2 = bN2; bC3 = bN3;
  }
  float* crow = Cm + (long long)(tm * 16 + half * 8) * ldc + col_off + tg * 64 + mi;
  #pragma unroll
  for (int r = 0; r < 8; r++) {
    crow[(long long)r * ldc + 0]  = acc0[r];
    crow[(long long)r * ldc + 16] = acc1[r];
    crow[(long long)r * ldc + 32] = acc2[r];
    crow[(long long)r * ldc + 48] = acc3[r];
  }
}

// head: +b1, LN, lrelu in-place on [NT,64]
__global__ void k_head_ln(float* __restrict__ hb, const float* __restrict__ b1,
                          const float* __restrict__ g, const float* __restrict__ b) {
  int row = blockIdx.x, c = threadIdx.x;
  __shared__ float s1[64], s2[64], red[2];
  float v = hb[(long long)row * 64 + c] + b1[c];
  ln_stats(v, c, s1, s2, red);
  float y = (v - red[0]) * red[1] * g[c] + b[c];
  hb[(long long)row * 64 + c] = y > 0.f ? y : 0.01f * y;
}
// v = h@w2 + b2 ; mean-pool over batches via atomics
__global__ void k_head_v(const float* __restrict__ hb, const float* __restrict__ w2,
                         const float* __restrict__ b2, const int* __restrict__ batch,
                         float* __restrict__ vsum, float* __restrict__ cnt, int N) {
  int i = blockIdx.x * blockDim.x + threadIdx.x;
  if (i >= N) return;
  float s = b2[0];
  #pragma unroll
  for (int k = 0; k < 64; k++) s += hb[(long long)i * 64 + k] * w2[k];
  atomicAdd(&vsum[batch[i]], s);
  atomicAdd(&cnt[batch[i]], 1.f);
}
__global__ void k_v_final(const float* vsum, const float* cnt, float* out) {
  int i = blockIdx.x * blockDim.x + threadIdx.x;
  if (i < 100) out[i] = vsum[i] / cnt[i];
}
// actor heads: only 100 rows -> tiny VALU kernel (no GEMM needed)
__global__ void k_actor_head(const float* __restrict__ xcat, const int* __restrict__ ptr,
                             const float* __restrict__ w1, const float* __restrict__ b1,
                             const float* __restrict__ g, const float* __restrict__ bb,
                             const float* __restrict__ w2, const float* __restrict__ b2,
                             float* __restrict__ out, int dout) {
  int i = blockIdx.x, c = threadIdx.x;            // 100 blocks x 64 threads
  __shared__ float s1[64], s2[64], red[2], hs[64];
  const float* z = xcat + (long long)ptr[i] * 256;
  float s = b1[c];
  for (int k = 0; k < 256; k++) s += z[k] * w1[k * 64 + c];
  ln_stats(s, c, s1, s2, red);
  float y = (s - red[0]) * red[1] * g[c] + bb[c];
  y = y > 0.f ? y : 0.01f * y;
  hs[c] = y; __syncthreads();
  if (c < dout) {
    float o = b2[c];
    #pragma unroll
    for (int k = 0; k < 64; k++) o += hs[k] * w2[k * dout + c];
    out[i * dout + c] = o;
  }
}

static inline unsigned cdivu(long long a, long long b) { return (unsigned)((a + b - 1) / b); }

extern "C" void kernel_launch(void* const* d_in, const int* in_sizes, int n_in,
                              void* d_out, int out_size, void* d_ws, size_t ws_size,
                              hipStream_t stream) {
  (void)in_sizes; (void)n_in; (void)out_size;
  // Param leaves flattened jax-style (dict keys sorted, recursively).
  // hetero block (30 leaves): +0 g_dt{att_dst,att_edge,att_src,bias,w_dst,w_edge,w_res,w_src}
  //   +8 g_tt{...}  +16 g_vt{...}  +24 ln1_b +25 ln1_g +26 ln2_b +27 ln2_g +28 ln4_b +29 ln4_g
  // head block (6 leaves): b1,b2,ln_b,ln_g,w1,w2
  const int ACTOR = 0, ACTOR_D = 30, ACTOR_P = 36, CRITIC = 42, CRITIC_HEAD = 72;
  const int IN_TASKS = 78, IN_DATA = 79, IN_DEV = 80, IN_EIDT = 81, IN_EADT = 82,
            IN_EIVT = 83, IN_EAVT = 84, IN_EITT = 85, IN_EATT = 86, IN_BATCH = 87, IN_PTR = 88;
  auto F  = [&](int i) { return (const float*)d_in[i]; };
  auto Ip = [&](int i) { return (const int*)d_in[i]; };

  // -------- workspace carve --------
  char* wsb = (char*)d_ws;
  size_t off = 0;
  auto carve = [&](size_t bytes) -> char* {
    char* p = wsb + off; off = (off + bytes + 255) & ~(size_t)255; return p;
  };
  float* XCAT = (float*)carve((size_t)kNT * 256 * 4);           // concat output [NT,256]
  char* XSraw = carve((size_t)kNT * 128 * 4);                   // union: xs f32 / xcat bf16
  float* XS = (float*)XSraw;
  unsigned short* XCATBF = (unsigned short*)XSraw;
  float* OUTACC = (float*)carve((size_t)kNT * 128 * 4);
  float* RES = (float*)carve((size_t)kNT * 64 * 4);             // residual / head hidden
  unsigned short* TBF = (unsigned short*)carve((size_t)kNT * 64 * 2);
  unsigned short* DBF = (unsigned short*)carve((size_t)kND * 32 * 2);
  unsigned short* VBF = (unsigned short*)carve((size_t)16 * 32 * 2);  // devices padded 16x32
  float* ASRC = (float*)carve((size_t)kNT * 2 * 4);
  float* ADST = (float*)carve((size_t)kNT * 2 * 4);
  unsigned* AMAX = (unsigned*)carve((size_t)kNT * 2 * 4);
  float* DEN = (float*)carve((size_t)kNT * 2 * 4);
  float* AE = (float*)carve((size_t)300000 * 2 * 4);
  float* ALPHA = (float*)carve((size_t)300000 * 2 * 4);
  unsigned short* WT1 = (unsigned short*)carve(128 * 64 * 2);   // w_srcT [128,Kpad]
  unsigned short* WT2 = (unsigned short*)carve(64 * 64 * 2);    // w_resT
  unsigned short* WT3 = (unsigned short*)carve(64 * 256 * 2);   // head w1T
  float* FOLD = (float*)carve(1024 * 4);
  float* WSA = FOLD, *WDA = FOLD + 128, *WEF = FOLD + 256, *SUM8 = FOLD + 288, *ALOOP = FOLD + 296;
  float* VSUM = (float*)carve(128 * 4);
  float* CNT = (float*)carve(128 * 4);
  if (off > ws_size) return;   // insufficient scratch

  const float* tasks_x = F(IN_TASKS);
  const float* data_x = F(IN_DATA);
  const float* dev_x = F(IN_DEV);
  const int* ei_dt = Ip(IN_EIDT); const float* ea_dt = F(IN_EADT);
  const int* ei_vt = Ip(IN_EIVT); const float* ea_vt = F(IN_EAVT);
  const int* ei_tt = Ip(IN_EITT); const float* ea_tt = F(IN_EATT);
  const int* batch = Ip(IN_BATCH); const int* ptr = Ip(IN_PTR);
  float* out_p = (float*)d_out;          // p_logits [100,8]
  float* out_d = out_p + 800;            // d_logits [100,8]
  float* out_v = out_p + 1600;           // v [100,1]

  dim3 T(256);
  // one-time bf16 activation staging
  k_cvt_bf16_pad<<<cdivu((long long)kNT * 64, 256), T, 0, stream>>>(tasks_x, TBF, kNT, 64, kNT, 64);
  k_cvt_bf16_pad<<<cdivu((long long)kND * 32, 256), T, 0, stream>>>(data_x, DBF, kND, 32, kND, 32);
  k_cvt_bf16_pad<<<cdivu(16 * 32, 256), T, 0, stream>>>(dev_x, VBF, 8, 16, 16, 32);

  // waves = (M/16) * (N/64), one 16x64 strip per wave
  auto launch_gemm = [&](const unsigned short* A, const unsigned short* BT, float* C,
                         int M, int K, int N, int ldc, int col_off) {
    long long waves = ((long long)M / 16) * (N / 64);
    k_gemm_bf16<<<cdivu(waves * 32, 256), T, 0, stream>>>(A, BT, C, M, K, N, ldc, col_off);
  };

  auto run_conv = [&](int gb, const float* srcx, const unsigned short* srcbf,
                      int ns, int mspad, int fsrc, int kpad,
                      const int* ei, const float* ea, int E, int Etot,
                      int lng, int lnb, int col_off) {
    const int* es = ei; const int* ed = ei + E;
    k_wt_bf16<<<cdivu(128 * kpad, 256), T, 0, stream>>>(F(gb + 7), WT1, fsrc, 128, kpad);
    k_wt_bf16<<<cdivu(64 * 64, 256), T, 0, stream>>>(F(gb + 6), WT2, 64, 64, 64);
    k_fold_att<<<1, 256, 0, stream>>>(F(gb + 7), F(gb + 2), WSA, fsrc);   // w_src . att_src
    k_fold_att<<<1, 256, 0, stream>>>(F(gb + 4), F(gb + 0), WDA, 64);     // w_dst . att_dst
    k_fold_att<<<1, 64, 0, stream>>>(F(gb + 5), F(gb + 1), WEF, 8);       // w_edge . att_edge
    k_fill_f32<<<cdivu((long long)kNT * 128, 256), T, 0, stream>>>(OUTACC, 0.f, (long long)kNT * 128);
    k_fill_f32<<<cdivu(kNT * 2, 256), T, 0, stream>>>(DEN, 0.f, kNT * 2);
    k_fill_u32<<<cdivu(kNT * 2, 256), T, 0, stream>>>(AMAX, 0u, kNT * 2);
    k_fill_f32<<<1, 32, 0, stream>>>(SUM8, 0.f, 10);
    launch_gemm(srcbf, WT1, XS, mspad, kpad, 128, 128, 0);    // xs = x_src @ w_src
    launch_gemm(TBF, WT2, RES, kNT, 64, 64, 64, 0);           // residual = tasks @ w_res
    k_att_nodes<<<cdivu(ns, 256), T, 0, stream>>>(srcx, WSA, ASRC, ns, fsrc);
    k_att_nodes<<<cdivu(kNT, 256), T, 0, stream>>>(tasks_x, WDA, ADST, kNT, 64);
    k_edge_ae<<<cdivu(E, 256), T, 0, stream>>>(ea, WEF, AE, E);
    if (Etot > E) {   // tt conv: self-loops use mean edge attrs
      k_colsum8<<<512, T, 0, stream>>>(ea, SUM8, E);
      k_loop_ae<<<1, 1, 0, stream>>>(SUM8, WEF, ALOOP, 1.f / (float)E);
    }
    k_edge_alpha<<<cdivu((long long)Etot * 2, 256), T, 0, stream>>>(es, ed, ASRC, ADST, AE, ALOOP, ALPHA, AMAX, E, Etot);
    k_edge_exp<<<cdivu((long long)Etot * 2, 256), T, 0, stream>>>(ed, ALPHA, AMAX, DEN, E, Etot);
    k_edge_scatter<<<cdivu((long long)Etot * 32, 256), T, 0, stream>>>(es, ed, ALPHA, DEN, XS, OUTACC, E, Etot);
    k_gat_epilogue<<<kNT, 64, 0, stream>>>(OUTACC, RES, F(gb + 3), F(lng), F(lnb), XCAT, col_off);
  };

  auto run_hetero = [&](int pb) {
    k_copy_tasks<<<cdivu((long long)kNT * 64, 256), T, 0, stream>>>(tasks_x, XCAT, (long long)kNT * 64);
    run_conv(pb + 0, data_x, DBF, kND, kND, 32, 32, ei_dt, ea_dt, kEDT, kEDT, pb + 25, pb + 24, 128); // a
    run_conv(pb + 16, dev_x, VBF, kNV, 16, 16, 32, ei_vt, ea_vt, kEVT, kEVT, pb + 27, pb + 26, 192);  // b
    run_conv(pb + 8, tasks_x, TBF, kNT, kNT, 64, 64, ei_tt, ea_tt, kETT, kETT + kNT, pb + 29, pb + 28, 64); // c
  };

  // ---- critic pass + value head ----
  run_hetero(CRITIC);
  k_cvt_bf16_pad<<<cdivu((long long)kNT * 256, 256), T, 0, stream>>>(XCAT, XCATBF, kNT, 256, kNT, 256);
  k_wt_bf16<<<cdivu(64 * 256, 256), T, 0, stream>>>(F(CRITIC_HEAD + 4), WT3, 256, 64, 256);
  launch_gemm(XCATBF, WT3, RES, kNT, 256, 64, 64, 0);
  k_head_ln<<<kNT, 64, 0, stream>>>(RES, F(CRITIC_HEAD + 0), F(CRITIC_HEAD + 3), F(CRITIC_HEAD + 2));
  k_fill_f32<<<1, 256, 0, stream>>>(VSUM, 0.f, 100);
  k_fill_f32<<<1, 256, 0, stream>>>(CNT, 0.f, 100);
  k_head_v<<<cdivu(kNT, 256), T, 0, stream>>>(RES, F(CRITIC_HEAD + 5), F(CRITIC_HEAD + 1), batch, VSUM, CNT, kNT);
  k_v_final<<<1, 128, 0, stream>>>(VSUM, CNT, out_v);

  // ---- actor pass + policy heads (only 100 rows needed) ----
  run_hetero(ACTOR);
  k_actor_head<<<100, 64, 0, stream>>>(XCAT, ptr, F(ACTOR_P + 4), F(ACTOR_P + 0), F(ACTOR_P + 3),
                                       F(ACTOR_P + 2), F(ACTOR_P + 5), F(ACTOR_P + 1), out_p, 8);
  k_actor_head<<<100, 64, 0, stream>>>(XCAT, ptr, F(ACTOR_D + 4), F(ACTOR_D + 0), F(ACTOR_D + 3),
                                       F(ACTOR_D + 2), F(ACTOR_D + 5), F(ACTOR_D + 1), out_d, 8);
}
